// Encoder_60610578482077
// MI455X (gfx1250) — matrix operand order repaired and checked by reading, not compiled, not run on hardware
//
#include <hip/hip_runtime.h>

// ---------------------------------------------------------------------------
// Types / helpers
// ---------------------------------------------------------------------------
typedef __attribute__((ext_vector_type(16))) __bf16 v16bf;
typedef __attribute__((ext_vector_type(8)))  float  v8f;

union BFV {
    v16bf v;
    unsigned u32[8];
};

__device__ __forceinline__ unsigned short f2bf(float f) {
    unsigned u = __float_as_uint(f);
    u += 0x7fffu + ((u >> 16) & 1u);   // round-to-nearest-even
    return (unsigned short)(u >> 16);
}

#define M_ROWS 16384   // B*S = 32*512
#define DMODEL 512
#define DFF    2048

// ---- CDNA5 data-movement feature detection (graceful fallback) ------------
#if defined(__AMDGCN__) && __has_builtin(__builtin_amdgcn_tensor_load_to_lds) && __has_builtin(__builtin_amdgcn_s_wait_tensorcnt)
#define HAVE_TDM 1
#endif
#if defined(__AMDGCN__) && __has_builtin(__builtin_amdgcn_global_load_async_to_lds_b128)
#define HAVE_ASYNC 1
#endif

typedef int v4i_ __attribute__((ext_vector_type(4)));
// Exact-pointee, target-address-space casts (same pattern HIP's atomic
// wrappers use for *1-typed builtins).
#define AS1_V4I(p) ((__attribute__((address_space(1))) v4i_*)(p))
#define AS3_V4I(p) ((__attribute__((address_space(3))) v4i_*)(p))

__device__ __forceinline__ void wait_async0() {
#if defined(__AMDGCN__)
#if __has_builtin(__builtin_amdgcn_s_wait_asynccnt)
    __builtin_amdgcn_s_wait_asynccnt(0);
#else
    asm volatile("s_wait_asynccnt 0x0" ::: "memory");
#endif
#endif
}

#if defined(HAVE_TDM)
typedef unsigned int v4u __attribute__((ext_vector_type(4)));
typedef int v8i __attribute__((ext_vector_type(8)));
typedef int v4i __attribute__((ext_vector_type(4)));

// Issue a 2D TDM load: tile (tile_w x tile_h) of 2-byte elements from a
// row-major tensor with row stride `stride_e` (elements) into LDS at lds_off.
// D# layout per CDNA5 ISA (group0 + group1, 2D tile).
__device__ __forceinline__ void tdm_load_2d_bf16(const void* gsrc, unsigned lds_off,
                                                 unsigned tile_w, unsigned tile_h,
                                                 unsigned tensor_w, unsigned tensor_h,
                                                 unsigned stride_e)
{
    unsigned long long ga = (unsigned long long)(size_t)gsrc;
    v4u g0;
    g0[0] = 0x1u;                                            // count=1 (valid user D#)
    g0[1] = lds_off;                                         // lds_addr (bytes)
    g0[2] = (unsigned)ga;                                    // global_addr[31:0]
    g0[3] = (unsigned)((ga >> 32) & 0x1ffffffu) | (2u << 30);// global_addr[56:32] | type=2
    v8i g1;
    g1[0] = (int)(1u << 16);                                 // data_size=1 (2 bytes)
    g1[1] = (int)((tensor_w & 0xffffu) << 16);               // tensor_dim0[15:0]
    g1[2] = (int)((tensor_w >> 16) | ((tensor_h & 0xffffu) << 16)); // td0[31:16]|td1[15:0]
    g1[3] = (int)((tensor_h >> 16) | (tile_w << 16));        // td1[31:16]|tile_dim0
    g1[4] = (int)(tile_h & 0xffffu);                         // tile_dim1 (tile_dim2=0)
    g1[5] = (int)stride_e;                                   // tensor_dim0_stride[31:0]
    g1[6] = 0;
    g1[7] = 0;
    v4i z4 = {0, 0, 0, 0};
    v8i z8 = {0, 0, 0, 0, 0, 0, 0, 0};
    __builtin_amdgcn_tensor_load_to_lds(g0, g1, z4, z4, z8, 0);
}
#endif

// ---------------------------------------------------------------------------
// Weight conversion kernels (f32 -> bf16, with layout fix for WQ/WK/WV)
// ---------------------------------------------------------------------------
__global__ void cvt_plain(const float* __restrict__ s, unsigned short* __restrict__ d, int n) {
    int i = blockIdx.x * blockDim.x + threadIdx.x;
    if (i < n) d[i] = f2bf(s[i]);
}

// [H=8, D=512, dk=64] -> [D=512, H*dk=512] row-major
__global__ void cvt_qkv(const float* __restrict__ s, unsigned short* __restrict__ d) {
    int i = blockIdx.x * blockDim.x + threadIdx.x;
    if (i >= 512 * 512) return;
    int col = i & 511;          // h*64 + k
    int row = i >> 9;           // D index
    int h = col >> 6, k = col & 63;
    d[i] = f2bf(s[(size_t)((h << 9) + row) * 64 + k]);
}

// ---------------------------------------------------------------------------
// Embedding gather + *sqrt(d_k)=8
// ---------------------------------------------------------------------------
__global__ void embed_kernel(const int* __restrict__ tok, const float* __restrict__ emb,
                             float* __restrict__ Xf, unsigned short* __restrict__ Xb, int total) {
    int i = blockIdx.x * blockDim.x + threadIdx.x;
    if (i >= total) return;
    int row = i >> 9, c = i & 511;
    float v = emb[(size_t)tok[row] * DMODEL + c] * 8.0f;
    Xf[i] = v;
    Xb[i] = f2bf(v);
}

// ---------------------------------------------------------------------------
// bf16 GEMM, f32 accumulate:  C[M,N] = A[M,K] * B[K,N] (+bias, relu, +resid)
// Block: 256 threads (8 waves as 4x2), tile 128x128, K-tile 32.
// A tile fed by the Tensor Data Mover (wave 0 issues, s_wait_tensorcnt).
// ---------------------------------------------------------------------------
__global__ __launch_bounds__(256)
void gemm_bf16(const unsigned short* __restrict__ A,
               const unsigned short* __restrict__ B,
               int M, int N, int K,
               const float* __restrict__ bias,
               const float* __restrict__ resid,
               float* __restrict__ Cf,
               unsigned short* __restrict__ Cb,
               int relu)
{
    __shared__ unsigned short At[128 * 32];   // [m][k]   (8 KB)
    __shared__ unsigned short Bt[128 * 32];   // [n][k] transposed (8 KB)

    const int tid  = threadIdx.x;
    const int wave = tid >> 5;
    const int lane = tid & 31;
    const int wm   = wave >> 1;     // 0..3 : row group of 32
    const int wn   = wave & 1;      // 0..1 : col group of 64
    const int lh   = lane >> 4;     // lane half
    const int ln   = lane & 15;
    const int n0   = blockIdx.x * 128;
    const int m0   = blockIdx.y * 128;

    v8f c[2][4];
#pragma unroll
    for (int mf = 0; mf < 2; ++mf)
#pragma unroll
        for (int f = 0; f < 4; ++f)
#pragma unroll
            for (int r = 0; r < 8; ++r) c[mf][f][r] = 0.0f;

    for (int kt = 0; kt < K; kt += 32) {
        __syncthreads();
        // ---- A tile: 128 rows x 32 cols bf16 --------------------------------
#if defined(HAVE_TDM)
        if (tid < 32) {   // one wave issues the DMA
            tdm_load_2d_bf16(A + (size_t)m0 * K + kt, (unsigned)(size_t)At,
                             /*tile_w=*/32, /*tile_h=*/128,
                             /*tensor_w=*/(unsigned)K, /*tensor_h=*/(unsigned)M,
                             /*stride_e=*/(unsigned)K);
        }
#else
#pragma unroll
        for (int it = 0; it < 2; ++it) {
            int slot = tid + it * 256;
            int r = slot >> 2, cc = (slot & 3) * 8;
            uint4 d = *(const uint4*)(A + (size_t)(m0 + r) * K + kt + cc);
            *(uint4*)(At + r * 32 + cc) = d;
        }
#endif
        // ---- B tile rows kt..kt+31, cols n0..n0+127 -> Bt[n][k] -------------
#pragma unroll
        for (int it = 0; it < 2; ++it) {
            int slot = tid + it * 256;
            int kr = slot >> 4, cc = (slot & 15) * 8;
            uint4 d = *(const uint4*)(B + (size_t)(kt + kr) * N + n0 + cc);
            unsigned short tmp[8];
            *(uint4*)tmp = d;
#pragma unroll
            for (int e = 0; e < 8; ++e) Bt[(cc + e) * 32 + kr] = tmp[e];
        }
#if defined(HAVE_TDM)
        if (tid < 32) __builtin_amdgcn_s_wait_tensorcnt(0);
#endif
        __syncthreads();

        // ---- compute: 2 A-frags x 4 B-frags per wave ------------------------
        BFV a[2];
#pragma unroll
        for (int mf = 0; mf < 2; ++mf) {
            int mr = wm * 32 + mf * 16 + ln;
#pragma unroll
            for (int i = 0; i < 8; ++i) {
                int kb = (i >> 2) * 16 + lh * 8 + (i & 3) * 2;
                a[mf].u32[i] = *(const unsigned*)(At + mr * 32 + kb);
            }
        }
#pragma unroll
        for (int f = 0; f < 4; ++f) {
            BFV b;
            int nc = wn * 64 + f * 16 + ln;
#pragma unroll
            for (int j = 0; j < 8; ++j) {
                int kb = (j >> 2) * 16 + lh * 8 + (j & 3) * 2;
                b.u32[j] = *(const unsigned*)(Bt + nc * 32 + kb);
            }
#pragma unroll
            for (int mf = 0; mf < 2; ++mf)
                c[mf][f] = __builtin_amdgcn_wmma_f32_16x16x32_bf16(
                    false, a[mf].v, false, b.v, (short)0, c[mf][f], false, false);
        }
    }

    // Epilogue. C layout: lane=(n%16)+16*(m/8), reg=m%8
#pragma unroll
    for (int mf = 0; mf < 2; ++mf)
#pragma unroll
        for (int f = 0; f < 4; ++f) {
            int col = n0 + wn * 64 + f * 16 + ln;
            float bs = bias ? bias[col] : 0.0f;
#pragma unroll
            for (int r = 0; r < 8; ++r) {
                int row = m0 + wm * 32 + mf * 16 + lh * 8 + r;
                float v = c[mf][f][r] + bs;
                if (relu) v = v > 0.0f ? v : 0.0f;
                if (resid) v += resid[(size_t)row * N + col];
                if (Cf) Cf[(size_t)row * N + col] = v;
                if (Cb) Cb[(size_t)row * N + col] = f2bf(v);
            }
        }
}

// ---------------------------------------------------------------------------
// Flash attention: per (b,h) with 64 query rows per block. S=512, d_k=d_v=64.
// Q/K/V/O are [M, 512] bf16, head h occupies cols h*64..h*64+63.
// Q/K tiles use async global->LDS (ASYNCcnt); V needs a transpose so it goes
// through registers.
// ---------------------------------------------------------------------------
__global__ __launch_bounds__(128)
void attn_kernel(const unsigned short* __restrict__ Q,
                 const unsigned short* __restrict__ K,
                 const unsigned short* __restrict__ V,
                 unsigned short* __restrict__ O)
{
    __shared__ unsigned short Qs[64 * 64];        // [q][k]
    __shared__ unsigned short Ks[64 * 64];        // [key][k]
    __shared__ unsigned short Vs[64 * 64];        // transposed [dv][key]
    __shared__ unsigned short Ps[4][16 * 64];     // per-wave P staging [m][key]

    const int tid  = threadIdx.x;
    const int wave = tid >> 5;
    const int lane = tid & 31;
    const int lh   = lane >> 4;
    const int ln   = lane & 15;

    const int q0 = blockIdx.x * 64;
    const int h  = blockIdx.y;
    const int b  = blockIdx.z;
    const size_t rowbase = (size_t)b * 512;
    const int hc = h * 64;

    // Load Q tile (async if available; waited before first use)
#pragma unroll
    for (int it = 0; it < 4; ++it) {
        int slot = tid + it * 128;
        int r = slot >> 3, cc = (slot & 7) * 8;
#if defined(HAVE_ASYNC)
        __builtin_amdgcn_global_load_async_to_lds_b128(
            AS1_V4I(Q + (rowbase + q0 + r) * DMODEL + hc + cc),
            AS3_V4I(Qs + r * 64 + cc), 0, 0);
#else
        *(uint4*)(Qs + r * 64 + cc) =
            *(const uint4*)(Q + (rowbase + q0 + r) * DMODEL + hc + cc);
#endif
    }

    float mrow[8], lrow[8];
    v8f o[4];
#pragma unroll
    for (int r = 0; r < 8; ++r) { mrow[r] = -1e30f; lrow[r] = 0.0f; }
#pragma unroll
    for (int f = 0; f < 4; ++f)
#pragma unroll
        for (int r = 0; r < 8; ++r) o[f][r] = 0.0f;

    for (int kb = 0; kb < 8; ++kb) {
        __syncthreads();
#pragma unroll
        for (int it = 0; it < 4; ++it) {
            int slot = tid + it * 128;
            int r = slot >> 3, cc = (slot & 7) * 8;
#if defined(HAVE_ASYNC)
            __builtin_amdgcn_global_load_async_to_lds_b128(
                AS1_V4I(K + (rowbase + kb * 64 + r) * DMODEL + hc + cc),
                AS3_V4I(Ks + r * 64 + cc), 0, 0);
#else
            *(uint4*)(Ks + r * 64 + cc) =
                *(const uint4*)(K + (rowbase + kb * 64 + r) * DMODEL + hc + cc);
#endif
            uint4 d = *(const uint4*)(V + (rowbase + kb * 64 + r) * DMODEL + hc + cc);
            unsigned short tmp[8];
            *(uint4*)tmp = d;
#pragma unroll
            for (int e = 0; e < 8; ++e) Vs[(cc + e) * 64 + r] = tmp[e];
        }
#if defined(HAVE_ASYNC)
        wait_async0();   // covers Q on first iteration and this K tile
#endif
        __syncthreads();

        // S = Q * K^T  (k-dim 64 -> 2 WMMA steps)
        v8f s[4];
#pragma unroll
        for (int f = 0; f < 4; ++f)
#pragma unroll
            for (int r = 0; r < 8; ++r) s[f][r] = 0.0f;

#pragma unroll
        for (int ks = 0; ks < 2; ++ks) {
            BFV a;
            int mr = wave * 16 + ln;
#pragma unroll
            for (int i = 0; i < 8; ++i) {
                int kk = ks * 32 + (i >> 2) * 16 + lh * 8 + (i & 3) * 2;
                a.u32[i] = *(const unsigned*)(Qs + mr * 64 + kk);
            }
#pragma unroll
            for (int f = 0; f < 4; ++f) {
                BFV bb;
                int nn = f * 16 + ln;   // key index
#pragma unroll
                for (int j = 0; j < 8; ++j) {
                    int kk = ks * 32 + (j >> 2) * 16 + lh * 8 + (j & 3) * 2;
                    bb.u32[j] = *(const unsigned*)(Ks + nn * 64 + kk);
                }
                s[f] = __builtin_amdgcn_wmma_f32_16x16x32_bf16(
                    false, a.v, false, bb.v, (short)0, s[f], false, false);
            }
        }

        // Online softmax (scale 1/sqrt(64)=0.125). Rows live in 16-lane halves.
#pragma unroll
        for (int r = 0; r < 8; ++r) {
            float mx = -1e30f;
#pragma unroll
            for (int f = 0; f < 4; ++f) {
                float v = s[f][r] * 0.125f;
                s[f][r] = v;
                mx = fmaxf(mx, v);
            }
#pragma unroll
            for (int d = 1; d < 16; d <<= 1) mx = fmaxf(mx, __shfl_xor(mx, d, 32));
            float nm = fmaxf(mrow[r], mx);
            float alpha = __expf(mrow[r] - nm);
            mrow[r] = nm;
            float rs = 0.0f;
#pragma unroll
            for (int f = 0; f < 4; ++f) {
                float p = __expf(s[f][r] - nm);
                s[f][r] = p;
                rs += p;
            }
#pragma unroll
            for (int d = 1; d < 16; d <<= 1) rs += __shfl_xor(rs, d, 32);
            lrow[r] = lrow[r] * alpha + rs;
#pragma unroll
            for (int f = 0; f < 4; ++f) o[f][r] *= alpha;
        }

        // Stage P (C layout -> row-major bf16) wave-privately; same-wave LDS is in-order.
#pragma unroll
        for (int f = 0; f < 4; ++f)
#pragma unroll
            for (int r = 0; r < 8; ++r)
                Ps[wave][(lh * 8 + r) * 64 + f * 16 + ln] = f2bf(s[f][r]);

        // O += P * V
#pragma unroll
        for (int ks = 0; ks < 2; ++ks) {
            BFV a;
#pragma unroll
            for (int i = 0; i < 8; ++i) {
                int kk = ks * 32 + (i >> 2) * 16 + lh * 8 + (i & 3) * 2;
                a.u32[i] = *(const unsigned*)(Ps[wave] + ln * 64 + kk);
            }
#pragma unroll
            for (int f = 0; f < 4; ++f) {
                BFV bb;
                int nn = f * 16 + ln;   // dv index
#pragma unroll
                for (int j = 0; j < 8; ++j) {
                    int kk = ks * 32 + (j >> 2) * 16 + lh * 8 + (j & 3) * 2;
                    bb.u32[j] = *(const unsigned*)(Vs + nn * 64 + kk);
                }
                o[f] = __builtin_amdgcn_wmma_f32_16x16x32_bf16(
                    false, a.v, false, bb.v, (short)0, o[f], false, false);
            }
        }
    }

    // Normalize and store (concat layout: col = h*64 + dv)
#pragma unroll
    for (int f = 0; f < 4; ++f)
#pragma unroll
        for (int r = 0; r < 8; ++r) {
            int row = q0 + wave * 16 + lh * 8 + r;
            O[(rowbase + row) * DMODEL + hc + f * 16 + ln] = f2bf(o[f][r] / lrow[r]);
        }
}

// ---------------------------------------------------------------------------
// Deterministic BatchNorm: partial sums -> reduce -> apply
// ---------------------------------------------------------------------------
__global__ __launch_bounds__(256)
void bn_partial(const float* __restrict__ Y, float* __restrict__ part) {
    int t = threadIdx.x;
    int r0 = blockIdx.x * 128;
    int c = t * 2;
    float s0 = 0, s1 = 0, q0 = 0, q1 = 0;
    for (int r = 0; r < 128; ++r) {
        float2 v = *(const float2*)(Y + (size_t)(r0 + r) * DMODEL + c);
        s0 += v.x; s1 += v.y;
        q0 += v.x * v.x; q1 += v.y * v.y;
    }
    float* p = part + (size_t)blockIdx.x * 1024;
    p[c] = s0; p[c + 1] = s1;
    p[512 + c] = q0; p[512 + c + 1] = q1;
}

__global__ void bn_reduce(const float* __restrict__ part, float* __restrict__ sums) {
    int i = blockIdx.x * blockDim.x + threadIdx.x;
    if (i >= 1024) return;
    float s = 0.0f;
    for (int b = 0; b < 128; ++b) s += part[(size_t)b * 1024 + i];
    sums[i] = s;
}

__global__ void bn_apply(const float* __restrict__ Y, const float* __restrict__ sums,
                         const float* __restrict__ g, const float* __restrict__ be,
                         float* __restrict__ Xf, unsigned short* __restrict__ Xb, int total) {
    int i = blockIdx.x * blockDim.x + threadIdx.x;
    if (i >= total) return;
    int c = i & 511;
    const float inv = 1.0f / (float)M_ROWS;
    float mean = sums[c] * inv;
    float var = sums[512 + c] * inv - mean * mean;
    float v = (Y[i] - mean) * rsqrtf(var + 1e-5f) * g[c] + be[c];
    Xf[i] = v;
    Xb[i] = f2bf(v);
}

// ---------------------------------------------------------------------------
// Host launcher
// ---------------------------------------------------------------------------
extern "C" void kernel_launch(void* const* d_in, const int* in_sizes, int n_in,
                              void* d_out, int out_size, void* d_ws, size_t ws_size,
                              hipStream_t stream) {
    (void)in_sizes; (void)n_in; (void)out_size; (void)ws_size;

    const int*   tok = (const int*)d_in[0];
    const float* emb = (const float*)d_in[1];
    const float* WQ  = (const float*)d_in[2];
    const float* WK  = (const float*)d_in[3];
    const float* WV  = (const float*)d_in[4];
    const float* WO  = (const float*)d_in[5];
    const float* g1  = (const float*)d_in[6];
    const float* be1 = (const float*)d_in[7];
    const float* W1  = (const float*)d_in[8];
    const float* b1  = (const float*)d_in[9];
    const float* W2  = (const float*)d_in[10];
    const float* b2  = (const float*)d_in[11];
    const float* g2  = (const float*)d_in[12];
    const float* be2 = (const float*)d_in[13];

    char* ws = (char*)d_ws;
    float*          Xf  = (float*)(ws + 0);                       // 33.5 MB
    float*          Yf  = (float*)(ws + 33554432);                // 33.5 MB
    unsigned short* Xb  = (unsigned short*)(ws + 67108864);       // 16.8 MB
    unsigned short* Qb  = (unsigned short*)(ws + 83886080);
    unsigned short* Kb  = (unsigned short*)(ws + 100663296);
    unsigned short* Vb  = (unsigned short*)(ws + 117440512);
    unsigned short* Cb  = (unsigned short*)(ws + 134217728);
    unsigned short* Hb  = Qb;  // FFN hidden aliases dead Q/K/V/Cat (67.1 MB)
    float*          sums = (float*)(ws + 150994944);
    float*          part = (float*)(ws + 151003136);              // 512 KB
    unsigned short* wqb = (unsigned short*)(ws + 151527424);
    unsigned short* wkb = (unsigned short*)(ws + 154673152);
    unsigned short* wvb = (unsigned short*)(ws + 157818880);
    unsigned short* wob = (unsigned short*)(ws + 160964608);
    unsigned short* w1b = (unsigned short*)(ws + 164110336);      // 12.6 MB
    unsigned short* w2b = (unsigned short*)(ws + 176693248);      // 12.6 MB

    const int total = M_ROWS * DMODEL;   // 8388608

    // Convert all weights to bf16 (with layout fix for WQ/WK/WV)
    for (int i = 0; i < 6; ++i) {
        cvt_qkv<<<1024, 256, 0, stream>>>(WQ + (size_t)i * 262144, wqb + (size_t)i * 262144);
        cvt_qkv<<<1024, 256, 0, stream>>>(WK + (size_t)i * 262144, wkb + (size_t)i * 262144);
        cvt_qkv<<<1024, 256, 0, stream>>>(WV + (size_t)i * 262144, wvb + (size_t)i * 262144);
        cvt_plain<<<1024, 256, 0, stream>>>(WO + (size_t)i * 262144, wob + (size_t)i * 262144, 262144);
        cvt_plain<<<4096, 256, 0, stream>>>(W1 + (size_t)i * 1048576, w1b + (size_t)i * 1048576, 1048576);
        cvt_plain<<<4096, 256, 0, stream>>>(W2 + (size_t)i * 1048576, w2b + (size_t)i * 1048576, 1048576);
    }

    // Embedding
    embed_kernel<<<total / 256, 256, 0, stream>>>(tok, emb, Xf, Xb, total);

    for (int i = 0; i < 6; ++i) {
        // Q/K/V projections: [16384,512] x [512,512] -> bf16
        gemm_bf16<<<dim3(4, 128), 256, 0, stream>>>(Xb, wqb + (size_t)i * 262144,
            M_ROWS, DMODEL, DMODEL, nullptr, nullptr, nullptr, Qb, 0);
        gemm_bf16<<<dim3(4, 128), 256, 0, stream>>>(Xb, wkb + (size_t)i * 262144,
            M_ROWS, DMODEL, DMODEL, nullptr, nullptr, nullptr, Kb, 0);
        gemm_bf16<<<dim3(4, 128), 256, 0, stream>>>(Xb, wvb + (size_t)i * 262144,
            M_ROWS, DMODEL, DMODEL, nullptr, nullptr, nullptr, Vb, 0);

        // Attention (flash) -> concat buffer
        attn_kernel<<<dim3(8, 8, 32), 128, 0, stream>>>(Qb, Kb, Vb, Cb);

        // Output projection + residual
        gemm_bf16<<<dim3(4, 128), 256, 0, stream>>>(Cb, wob + (size_t)i * 262144,
            M_ROWS, DMODEL, DMODEL, nullptr, Xf, Yf, nullptr, 0);

        // BatchNorm 1
        bn_partial<<<128, 256, 0, stream>>>(Yf, part);
        bn_reduce<<<4, 256, 0, stream>>>(part, sums);
        bn_apply<<<total / 256, 256, 0, stream>>>(Yf, sums,
            g1 + (size_t)i * 512, be1 + (size_t)i * 512, Xf, Xb, total);

        // FFN: relu(X W1 + b1) W2 + b2 + X
        gemm_bf16<<<dim3(16, 128), 256, 0, stream>>>(Xb, w1b + (size_t)i * 1048576,
            M_ROWS, DFF, DMODEL, b1 + (size_t)i * 2048, nullptr, nullptr, Hb, 1);
        gemm_bf16<<<dim3(4, 128), 256, 0, stream>>>(Hb, w2b + (size_t)i * 1048576,
            M_ROWS, DMODEL, DFF, b2 + (size_t)i * 512, Xf, Yf, nullptr, 0);

        // BatchNorm 2 (last layer writes final f32 output)
        bn_partial<<<128, 256, 0, stream>>>(Yf, part);
        bn_reduce<<<4, 256, 0, stream>>>(part, sums);
        float* xdst = (i == 5) ? (float*)d_out : Xf;
        bn_apply<<<total / 256, 256, 0, stream>>>(Yf, sums,
            g2 + (size_t)i * 512, be2 + (size_t)i * 512, xdst, Xb, total);
    }
}
